// Temporal_Attention_55276229099914
// MI455X (gfx1250) — matrix operand order repaired and checked by reading, not compiled, not run on hardware
//
#include <hip/hip_runtime.h>
#include <hip/hip_bf16.h>
#include <math.h>

// ---------------------------------------------------------------------------
// Transformer block (QKV proj -> causal attention over T -> FFN(GELU) ->
// residual -> LayerNorm) for x[16,48,207,512], 8 heads, d=64.
// MI455X (gfx1250, wave32) strategy:
//   * all matmuls via v_wmma_f32_16x16x32_bf16 (f32 accumulate)
//   * Q/K attention staging via global_load_async_to_lds_b128 (ASYNCcnt)
//   * weights pre-transposed so WMMA B-fragments are K-contiguous 16B loads
//   * GEMM epilogue staged through LDS for coalesced global stores
// ---------------------------------------------------------------------------

typedef __attribute__((ext_vector_type(16))) __bf16 bf16x16;
typedef __attribute__((ext_vector_type(8)))  float  f32x8;

#define F_DIM   512
#define B_DIM   16
#define T_DIM   48
#define N_VERT  207
#define KHEADS  8
#define D_HEAD  64
#define M_ROWS  (B_DIM * T_DIM * N_VERT)   // 158976, divisible by 64

static __device__ __forceinline__ unsigned short f2bf(float f) {
    union { float f; unsigned int u; } x; x.f = f;
    unsigned int r = x.u + 0x7FFFu + ((x.u >> 16) & 1u);   // RNE
    return (unsigned short)(r >> 16);
}

union FragU { bf16x16 v; uint4 q[2]; };

// Per-lane async copy of 16 bytes global -> LDS (gfx1250, ASYNCcnt tracked).
// ldsaddr = LDS byte address (low 32 bits of flat shared pointer).
static __device__ __forceinline__ void async_g2l_b128(unsigned ldsaddr,
                                                      const void* gaddr) {
    asm volatile("global_load_async_to_lds_b128 %0, %1, off"
                 :: "v"(ldsaddr), "v"((unsigned long long)(size_t)gaddr)
                 : "memory");
}
static __device__ __forceinline__ void wait_async0() {
    asm volatile("s_wait_asynccnt 0x0" ::: "memory");
}

// ----------------------------- f32 -> bf16 ---------------------------------
__global__ void cvt_f32_to_bf16(const float* __restrict__ src,
                                unsigned short* __restrict__ dst,
                                long long n4) {
    long long i = (long long)blockIdx.x * blockDim.x + threadIdx.x;
    if (i >= n4) return;
    float4 v = ((const float4*)src)[i];
    ushort4 o;
    o.x = f2bf(v.x); o.y = f2bf(v.y); o.z = f2bf(v.z); o.w = f2bf(v.w);
    ((ushort4*)dst)[i] = o;
}

// -------------------- weight transpose + bf16 convert ----------------------
// WT[n][k] = bf16(W[k][n]) so a WMMA B fragment (fixed N, K-contiguous)
// is two 16-byte reads per lane (K = g*8.. and 16+g*8..).
__global__ void wtrans_bf16(const float* __restrict__ w,
                            unsigned short* __restrict__ wt) {
    int idx = blockIdx.x * blockDim.x + threadIdx.x;   // 0..512*512-1
    int k = idx >> 9, n = idx & 511;
    wt[n * 512 + k] = f2bf(w[idx]);
}

// ------------------------------ WMMA GEMM ----------------------------------
// C[M,512] = A[M,512](bf16) * W[512,512] + bias, W given as WT (transposed).
// Block = 128 threads (4 waves); block tile 64x64; each wave: 32x32 via
// 2x2 accumulators of v_wmma_f32_16x16x32_bf16. K loop: 16 steps of 32.
// EPI: 0 = store bf16; 1 = exact GELU then bf16; 2 = + residual, store f32.
// Output staged through LDS for coalesced 16B global stores.
template <int EPI>
__global__ __launch_bounds__(128) void gemm_bf16(
    const unsigned short* __restrict__ A,
    const unsigned short* __restrict__ WT,
    const float* __restrict__ bias,
    unsigned short* __restrict__ outb,
    float* __restrict__ outf,
    const float* __restrict__ resid)
{
    __shared__ __align__(16) unsigned char smem[64 * 64 * 4];  // 16KB stage

    const int tid  = threadIdx.x;
    const int wave = tid >> 5, lane = tid & 31;
    const int lr   = lane & 15, g = lane >> 4;
    const int wr   = (wave >> 1) * 32, wc = (wave & 1) * 32;
    const int row0 = blockIdx.y * 64, col0 = blockIdx.x * 64;
    const int m0   = row0 + wr;
    const int n0   = col0 + wc;

    f32x8 acc[2][2];
    for (int nt = 0; nt < 2; ++nt) {
        float bv = bias[n0 + nt * 16 + lr];       // C col is lane-fixed
        f32x8 c; for (int i = 0; i < 8; ++i) c[i] = bv;
        acc[0][nt] = c; acc[1][nt] = c;
    }

    for (int ks = 0; ks < 16; ++ks) {
        const int k0 = ks * 32;
        FragU a[2], b[2];
        for (int mt = 0; mt < 2; ++mt) {
            const unsigned short* ar =
                A + (size_t)(m0 + mt * 16 + lr) * F_DIM + k0 + g * 8;
            a[mt].q[0] = *(const uint4*)(ar);        // e0..7 : K=g*8+e
            a[mt].q[1] = *(const uint4*)(ar + 16);   // e8..15: K=16+g*8+(e-8)
            if (ks < 15) __builtin_prefetch((const void*)(ar + 32), 0, 2);
        }
        for (int nt = 0; nt < 2; ++nt) {
            const unsigned short* br =
                WT + (size_t)(n0 + nt * 16 + lr) * F_DIM + k0 + g * 8;
            b[nt].q[0] = *(const uint4*)(br);        // e0..7: K=g*8+e, e8..15: K=16+g*8+(e-8), N=lane%16
            b[nt].q[1] = *(const uint4*)(br + 16);
        }
        for (int mt = 0; mt < 2; ++mt)
            for (int nt = 0; nt < 2; ++nt)
                acc[mt][nt] = __builtin_amdgcn_wmma_f32_16x16x32_bf16(
                    false, a[mt].v, false, b[nt].v, (short)0, acc[mt][nt],
                    false, false);
    }

    // ---- stage block tile to LDS (C layout: M = v + 8g, N = lane%16) ----
    if (EPI < 2) {
        unsigned short* Ot = (unsigned short*)smem;     // [64][64] bf16
        for (int mt = 0; mt < 2; ++mt)
            for (int nt = 0; nt < 2; ++nt)
                for (int v = 0; v < 8; ++v) {
                    int lrow = wr + mt * 16 + v + 8 * g;
                    int lcol = wc + nt * 16 + lr;
                    float val = acc[mt][nt][v];
                    if (EPI == 1)
                        val = 0.5f * val * (1.0f + erff(val * 0.70710678f));
                    Ot[lrow * 64 + lcol] = f2bf(val);
                }
        __syncthreads();
        for (int idx = tid; idx < 64 * 8; idx += 128) {  // 8 x uint4 per row
            int r = idx >> 3, c = idx & 7;
            *(uint4*)(outb + (size_t)(row0 + r) * F_DIM + col0 + c * 8) =
                ((const uint4*)Ot)[idx];
        }
    } else {
        float* Of = (float*)smem;                        // [64][64] f32
        for (int mt = 0; mt < 2; ++mt)
            for (int nt = 0; nt < 2; ++nt)
                for (int v = 0; v < 8; ++v) {
                    int lrow = wr + mt * 16 + v + 8 * g;
                    int lcol = wc + nt * 16 + lr;
                    Of[lrow * 64 + lcol] = acc[mt][nt][v];
                }
        __syncthreads();
        for (int idx = tid; idx < 64 * 16; idx += 128) { // 16 x float4 per row
            int r = idx >> 4, c = idx & 15;
            size_t o = (size_t)(row0 + r) * F_DIM + col0 + c * 4;
            float4 a4 = ((const float4*)Of)[idx];
            float4 r4 = *(const float4*)(resid + o);
            a4.x += r4.x; a4.y += r4.y; a4.z += r4.z; a4.w += r4.w;
            *(float4*)(outf + o) = a4;
        }
    }
}

// ------------------------------ attention ----------------------------------
// One 96-thread (3-wave) block per (n, b, head). Q,K staged into LDS with
// async global->LDS b128 copies; V transposed + zero-padded (s=48..63) in
// LDS. S = Q K^T/8 with causal mask (-32767), fp32 softmax through LDS,
// O = P V, bf16 out.
__global__ __launch_bounds__(96) void attn_kernel(
    const unsigned short* __restrict__ qb,
    const unsigned short* __restrict__ kb,
    const unsigned short* __restrict__ vb,
    unsigned short* __restrict__ ob)
{
    __shared__ __align__(16) unsigned short Qs[T_DIM * D_HEAD];
    __shared__ __align__(16) unsigned short Ks[T_DIM * D_HEAD];
    __shared__ __align__(16) unsigned short VT[D_HEAD * 64];   // [d][s pad 64]
    __shared__ __align__(16) unsigned short Pp[T_DIM * 64];    // [t][s pad 64]
    __shared__ float Sf[T_DIM * T_DIM];

    const int tid = threadIdx.x;
    const int nv = blockIdx.x, b = blockIdx.y, kh = blockIdx.z;
    const size_t hoff = (size_t)kh * D_HEAD;

    // stage Q,K via async copies: 48 rows x 64 bf16 = 384 x 16B chunks
    for (int idx = tid; idx < T_DIM * 8; idx += 96) {   // 4 full-EXEC iters
        int t = idx >> 3, c = idx & 7;
        size_t m = ((size_t)(b * T_DIM + t) * N_VERT + nv) * F_DIM + hoff + c * 8;
        async_g2l_b128((unsigned)(size_t)&Qs[idx * 8], qb + m);
        async_g2l_b128((unsigned)(size_t)&Ks[idx * 8], kb + m);
    }
    // stage V transposed, zero-pad s = 48..63 (scatter: regular ds stores)
    for (int idx = tid; idx < 64 * 64; idx += 96) {
        int s = idx >> 6, dd = idx & 63;
        unsigned short val = 0;
        if (s < T_DIM)
            val = vb[((size_t)(b * T_DIM + s) * N_VERT + nv) * F_DIM + hoff + dd];
        VT[dd * 64 + s] = val;
    }
    wait_async0();       // own async copies landed in LDS
    __syncthreads();     // everyone else's too

    const int wave = tid >> 5, lane = tid & 31, lr = lane & 15, g = lane >> 4;
    const int r0 = wave * 16;   // this wave's 16 query rows

    // S = Q K^T : 3 N-tiles, K = 64 in 2 steps
    f32x8 sacc[3];
    for (int nt = 0; nt < 3; ++nt)
        for (int i = 0; i < 8; ++i) sacc[nt][i] = 0.0f;
    for (int ks = 0; ks < 2; ++ks) {
        FragU a;
        const unsigned short* ar = Qs + (r0 + lr) * 64 + ks * 32 + g * 8;
        a.q[0] = *(const uint4*)(ar);
        a.q[1] = *(const uint4*)(ar + 16);
        for (int nt = 0; nt < 3; ++nt) {
            FragU bf_;
            const unsigned short* br = Ks + (nt * 16 + lr) * 64 + ks * 32 + g * 8;
            bf_.q[0] = *(const uint4*)(br);
            bf_.q[1] = *(const uint4*)(br + 16);
            sacc[nt] = __builtin_amdgcn_wmma_f32_16x16x32_bf16(
                false, a.v, false, bf_.v, (short)0, sacc[nt], false, false);
        }
    }
    // scale 1/sqrt(64), causal mask, spill to LDS f32
    for (int nt = 0; nt < 3; ++nt)
        for (int v = 0; v < 8; ++v) {
            int row = r0 + v + 8 * g;
            int col = nt * 16 + lr;
            float sv = sacc[nt][v] * 0.125f;
            if (col > row) sv = -32767.0f;
            Sf[row * T_DIM + col] = sv;
        }
    __syncthreads();

    // fp32 softmax, one thread per row; write bf16 P (zero-padded to K=64)
    if (tid < T_DIM) {
        float mx = -3.4e38f;
        for (int s = 0; s < T_DIM; ++s) mx = fmaxf(mx, Sf[tid * T_DIM + s]);
        float sum = 0.0f;
        for (int s = 0; s < T_DIM; ++s) sum += expf(Sf[tid * T_DIM + s] - mx);
        float inv = 1.0f / sum;
        for (int s = 0; s < T_DIM; ++s)
            Pp[tid * 64 + s] = f2bf(expf(Sf[tid * T_DIM + s] - mx) * inv);
        for (int s = T_DIM; s < 64; ++s) Pp[tid * 64 + s] = 0;
    }
    __syncthreads();

    // O = P V : 4 N-tiles of d, K = 64 (padded) in 2 steps
    f32x8 oacc[4];
    for (int nt = 0; nt < 4; ++nt)
        for (int i = 0; i < 8; ++i) oacc[nt][i] = 0.0f;
    for (int ks = 0; ks < 2; ++ks) {
        FragU a;
        const unsigned short* ar = Pp + (r0 + lr) * 64 + ks * 32 + g * 8;
        a.q[0] = *(const uint4*)(ar);
        a.q[1] = *(const uint4*)(ar + 16);
        for (int nt = 0; nt < 4; ++nt) {
            FragU bf_;
            const unsigned short* br = VT + (nt * 16 + lr) * 64 + ks * 32 + g * 8;
            bf_.q[0] = *(const uint4*)(br);
            bf_.q[1] = *(const uint4*)(br + 16);
            oacc[nt] = __builtin_amdgcn_wmma_f32_16x16x32_bf16(
                false, a.v, false, bf_.v, (short)0, oacc[nt], false, false);
        }
    }
    for (int nt = 0; nt < 4; ++nt)
        for (int v = 0; v < 8; ++v) {
            int t  = r0 + v + 8 * g;
            int dd = nt * 16 + lr;
            ob[((size_t)(b * T_DIM + t) * N_VERT + nv) * F_DIM + hoff + dd] =
                f2bf(oacc[nt][v]);
        }
}

// ------------------------------ LayerNorm ----------------------------------
__global__ __launch_bounds__(128) void ln_kernel(float* __restrict__ h,
                                                 const float* __restrict__ gamma,
                                                 const float* __restrict__ beta)
{
    const int tid = threadIdx.x;
    float* row = h + (size_t)blockIdx.x * F_DIM;
    float v[4]; float s = 0.0f, s2 = 0.0f;
    for (int i = 0; i < 4; ++i) {
        v[i] = row[tid + i * 128];
        s += v[i]; s2 += v[i] * v[i];
    }
    for (int off = 16; off > 0; off >>= 1) {
        s  += __shfl_down(s,  off, 32);
        s2 += __shfl_down(s2, off, 32);
    }
    __shared__ float rs[4], rs2[4];
    int wave = tid >> 5, lane = tid & 31;
    if (lane == 0) { rs[wave] = s; rs2[wave] = s2; }
    __syncthreads();
    if (tid == 0) {
        float a = 0.0f, c = 0.0f;
        for (int i = 0; i < 4; ++i) { a += rs[i]; c += rs2[i]; }
        rs[0] = a; rs2[0] = c;
    }
    __syncthreads();
    float mean = rs[0] * (1.0f / F_DIM);
    float var  = rs2[0] * (1.0f / F_DIM) - mean * mean;
    float rstd = rsqrtf(var + 1e-5f);
    for (int i = 0; i < 4; ++i) {
        int c = tid + i * 128;
        row[c] = gamma[c] * (v[i] - mean) * rstd + beta[c];
    }
}

// ------------------------------- launcher ----------------------------------
extern "C" void kernel_launch(void* const* d_in, const int* in_sizes, int n_in,
                              void* d_out, int out_size, void* d_ws, size_t ws_size,
                              hipStream_t stream) {
    const float* x     = (const float*)d_in[0];
    const float* Wq    = (const float*)d_in[1];
    const float* bq    = (const float*)d_in[2];
    const float* Wk    = (const float*)d_in[3];
    const float* bk    = (const float*)d_in[4];
    const float* Wv    = (const float*)d_in[5];
    const float* bv    = (const float*)d_in[6];
    const float* W1    = (const float*)d_in[7];
    const float* b1    = (const float*)d_in[8];
    const float* W2    = (const float*)d_in[9];
    const float* b2    = (const float*)d_in[10];
    const float* gamma = (const float*)d_in[11];
    const float* beta  = (const float*)d_in[12];
    float* out = (float*)d_out;

    const size_t ME = (size_t)M_ROWS * F_DIM;   // 81,395,712 elements
    unsigned short* ws  = (unsigned short*)d_ws;
    unsigned short* xb  = ws;             // bf16(x); reused as attention O
    unsigned short* qb  = xb + ME;        // Q; reused as H1 (post-GELU)
    unsigned short* kb  = qb + ME;
    unsigned short* vb  = kb + ME;
    unsigned short* wt  = vb + ME;        // 5 transposed bf16 weights
    unsigned short* wtq = wt;
    unsigned short* wtk = wt + 262144;
    unsigned short* wtv = wt + 2 * 262144;
    unsigned short* wt1 = wt + 3 * 262144;
    unsigned short* wt2 = wt + 4 * 262144;

    // x -> bf16
    cvt_f32_to_bf16<<<(unsigned)((ME / 4 + 255) / 256), 256, 0, stream>>>(
        x, xb, (long long)(ME / 4));
    // weights -> bf16 transposed
    wtrans_bf16<<<1024, 256, 0, stream>>>(Wq, wtq);
    wtrans_bf16<<<1024, 256, 0, stream>>>(Wk, wtk);
    wtrans_bf16<<<1024, 256, 0, stream>>>(Wv, wtv);
    wtrans_bf16<<<1024, 256, 0, stream>>>(W1, wt1);
    wtrans_bf16<<<1024, 256, 0, stream>>>(W2, wt2);

    dim3 gg(F_DIM / 64, M_ROWS / 64);   // (8, 2484)
    gemm_bf16<0><<<gg, 128, 0, stream>>>(xb, wtq, bq, qb, nullptr, nullptr);
    gemm_bf16<0><<<gg, 128, 0, stream>>>(xb, wtk, bk, kb, nullptr, nullptr);
    gemm_bf16<0><<<gg, 128, 0, stream>>>(xb, wtv, bv, vb, nullptr, nullptr);

    attn_kernel<<<dim3(N_VERT, B_DIM, KHEADS), 96, 0, stream>>>(qb, kb, vb, xb);

    gemm_bf16<1><<<gg, 128, 0, stream>>>(xb, wt1, b1, qb, nullptr, nullptr);
    gemm_bf16<2><<<gg, 128, 0, stream>>>(qb, wt2, b2, nullptr, out, x);

    ln_kernel<<<M_ROWS, 128, 0, stream>>>(out, gamma, beta);
}
